// EdgeWeightGNN_1090921693652
// MI455X (gfx1250) — compile-verified, matrix operand order checked by reading, not run on hardware
//
#include <hip/hip_runtime.h>

typedef __attribute__((ext_vector_type(16))) _Float16 v16h;
typedef __attribute__((ext_vector_type(8)))  _Float16 v8h;
typedef __attribute__((ext_vector_type(8)))  float    v8f;

#define WROW 136   // padded f16 row stride (halves) for weight matrices: 272B, 16B-aligned

// Build a v16h WMMA fragment from a 16-bit row in memory.
// ISA 16-bit A 16x32 layout: lane&15 = matrix row; lane>>4 (hi) selects K halves:
//   halves 0..7  = row[off + hi*8 + 0..7]       (K = hi*8 + 0..7   of this 32-chunk)
//   halves 8..15 = row[off + hi*8 + 16 + 0..7]  (K = hi*8 + 16..23 of this 32-chunk)
// B uses the symmetric layout (lane&15 = column), so the same loader works on
// row-major weight rows (B[k][n] = W[n][k]).
__device__ __forceinline__ v16h frag_f16(const _Float16* __restrict__ row, int off, int hi) {
  const _Float16* p = row + off + hi * 8;
  v8h lo = *(const v8h*)(p);
  v8h hh = *(const v8h*)(p + 16);
  v16h a;
#pragma unroll
  for (int t = 0; t < 8; ++t) { a[t] = lo[t]; a[t + 8] = hh[t]; }
  return a;
}

__device__ __forceinline__ v16h frag_f32(const float* __restrict__ row, int off, int hi) {
  const float* p = row + off + hi * 8;
  float4 a0 = *(const float4*)(p);
  float4 a1 = *(const float4*)(p + 4);
  float4 b0 = *(const float4*)(p + 16);
  float4 b1 = *(const float4*)(p + 20);
  v16h a;
  a[0]  = (_Float16)a0.x; a[1]  = (_Float16)a0.y; a[2]  = (_Float16)a0.z; a[3]  = (_Float16)a0.w;
  a[4]  = (_Float16)a1.x; a[5]  = (_Float16)a1.y; a[6]  = (_Float16)a1.z; a[7]  = (_Float16)a1.w;
  a[8]  = (_Float16)b0.x; a[9]  = (_Float16)b0.y; a[10] = (_Float16)b0.z; a[11] = (_Float16)b0.w;
  a[12] = (_Float16)b1.x; a[13] = (_Float16)b1.y; a[14] = (_Float16)b1.z; a[15] = (_Float16)b1.w;
  return a;
}

// ---------------- init: zero atomic accumulators, write scalar outputs -------
__global__ void k_init(float* __restrict__ agg0, float* __restrict__ agg1,
                       int n, float* __restrict__ out, long long e2) {
  long long t = (long long)blockIdx.x * blockDim.x + threadIdx.x;
  long long n4 = (long long)n * 4, n64 = (long long)n * 64;
  if (t < n4)  agg0[t] = 0.0f;
  if (t < n64) agg1[t] = 0.0f;
  if (t == 0) {
    out[3 * e2]     = (float)n;                 // num_real_nodes
    out[3 * e2 + 1] = (float)(2 * n + (n & 1)); // num_boundary_nodes
  }
}

// ---------------- prep: f32 -> f16 weight repack (padded rows) ---------------
__global__ void k_prep(const float* __restrict__ w1, const float* __restrict__ root1,
                       const float* __restrict__ rel1,
                       _Float16* __restrict__ W1h, _Float16* __restrict__ Wg) {
  int t = blockIdx.x * blockDim.x + threadIdx.x;
  if (t >= 64 * WROW) return;
  int o = t / WROW, k = t % WROW;
  W1h[t] = (k < 129) ? (_Float16)w1[o * 129 + k] : (_Float16)0.0f;
  float g = 0.0f;
  if (k < 64)       g = root1[o * 64 + k];
  else if (k < 128) g = rel1[o * 64 + (k - 64)];
  Wg[t] = (_Float16)g;
}

// ---------------- GC0: edge scatter (FEAT=4) ---------------------------------
__global__ void k_gc0_scatter(const float* __restrict__ x, const int* __restrict__ src,
                              const int* __restrict__ tgt, const float* __restrict__ ew,
                              float* __restrict__ agg0, int E) {
  int e = blockIdx.x * blockDim.x + threadIdx.x;
  if (e >= E) return;
  int s = src[e], d = tgt[e];
  float w = ew[e];
  float4 xv = *(const float4*)(x + (size_t)s * 4);
  atomicAdd(&agg0[(size_t)d * 4 + 0], w * xv.x);
  atomicAdd(&agg0[(size_t)d * 4 + 1], w * xv.y);
  atomicAdd(&agg0[(size_t)d * 4 + 2], w * xv.z);
  atomicAdd(&agg0[(size_t)d * 4 + 3], w * xv.w);
}

// ---------------- GC0: node update (4->64, scalar VALU) ----------------------
__global__ void k_gc0_node(const float* __restrict__ x, const float* __restrict__ agg0,
                           const float* __restrict__ wroot, const float* __restrict__ wrel,
                           const float* __restrict__ bias, float* __restrict__ h0, int n) {
  long long t = (long long)blockIdx.x * blockDim.x + threadIdx.x;
  if (t >= (long long)n * 64) return;
  int node = (int)(t >> 6), o = (int)(t & 63);
  float4 xv = *(const float4*)(x + (size_t)node * 4);
  float4 av = *(const float4*)(agg0 + (size_t)node * 4);
  float4 wr = *(const float4*)(wroot + o * 4);
  float4 wl = *(const float4*)(wrel + o * 4);
  float acc = bias[o]
            + wr.x * xv.x + wr.y * xv.y + wr.z * xv.z + wr.w * xv.w
            + wl.x * av.x + wl.y * av.y + wl.z * av.z + wl.w * av.w;
  h0[t] = fmaxf(acc, 0.0f);
}

// ---------------- GC1: edge scatter (64 feats, 2 per thread, f32 atomics) ----
__global__ void k_gc1_scatter(const float* __restrict__ h0, const int* __restrict__ src,
                              const int* __restrict__ tgt, const float* __restrict__ ew,
                              float* __restrict__ agg1, int E) {
  long long t = (long long)blockIdx.x * blockDim.x + threadIdx.x;
  if (t >= (long long)E * 32) return;
  int e = (int)(t >> 5), f2 = (int)(t & 31);          // one wave == one edge
  float w = ew[e];
  float2 hv = *(const float2*)(h0 + (size_t)src[e] * 64 + f2 * 2);
  float* dst = &agg1[(size_t)tgt[e] * 64 + f2 * 2];
  atomicAdd(dst,     w * hv.x);
  atomicAdd(dst + 1, w * hv.y);
}

// ---------------- GC1: node update via WMMA: [h0|agg1](16x128) @ Wg^T --------
__global__ void __launch_bounds__(32)
k_gc1_node(const float* __restrict__ h0, const float* __restrict__ agg1,
           const _Float16* __restrict__ Wg, const float* __restrict__ bias,
           _Float16* __restrict__ h1h, int n) {
  int lane = threadIdx.x;
  int c0 = lane & 15, hi = lane >> 4, rowbase = hi * 8;
  int ntiles = (n + 15) >> 4;

  v16h bf[4][4];
#pragma unroll
  for (int nt = 0; nt < 4; ++nt)
#pragma unroll
    for (int kc = 0; kc < 4; ++kc)
      bf[nt][kc] = frag_f16(Wg + (size_t)(nt * 16 + c0) * WROW, kc * 32, hi);
  float bv[4];
#pragma unroll
  for (int nt = 0; nt < 4; ++nt) bv[nt] = bias[nt * 16 + c0];

  for (int tile = blockIdx.x; tile < ntiles; tile += gridDim.x) {
    bool full = (tile * 16 + 16 <= n);        // wave-uniform: no exec toggles
    int node = tile * 16 + c0;
    int nodec = node < n ? node : n - 1;      // clamp: WMMA needs all lanes live
    const float* r0 = h0   + (size_t)nodec * 64;
    const float* r1 = agg1 + (size_t)nodec * 64;
    v16h a0 = frag_f32(r0, 0, hi);
    v16h a1 = frag_f32(r0, 32, hi);
    v16h a2 = frag_f32(r1, 0, hi);
    v16h a3 = frag_f32(r1, 32, hi);
#pragma unroll
    for (int nt = 0; nt < 4; ++nt) {
      v8f acc;
#pragma unroll
      for (int r = 0; r < 8; ++r) acc[r] = bv[nt];
      acc = __builtin_amdgcn_wmma_f32_16x16x32_f16(false, a0, false, bf[nt][0], (short)0, acc, false, false);
      acc = __builtin_amdgcn_wmma_f32_16x16x32_f16(false, a1, false, bf[nt][1], (short)0, acc, false, false);
      acc = __builtin_amdgcn_wmma_f32_16x16x32_f16(false, a2, false, bf[nt][2], (short)0, acc, false, false);
      acc = __builtin_amdgcn_wmma_f32_16x16x32_f16(false, a3, false, bf[nt][3], (short)0, acc, false, false);
      int col = nt * 16 + c0;
      _Float16* dst = h1h + (size_t)(tile * 16 + rowbase) * 64 + col;
      if (full) {
#pragma unroll
        for (int r = 0; r < 8; ++r) dst[(size_t)r * 64] = (_Float16)fmaxf(acc[r], 0.0f);
      } else {
#pragma unroll
        for (int r = 0; r < 8; ++r) {
          int nr = tile * 16 + rowbase + r;
          if (nr < n) h1h[(size_t)nr * 64 + col] = (_Float16)fmaxf(acc[r], 0.0f);
        }
      }
    }
  }
}

// ---------------- Edge MLP via WMMA: emb(16x129) @ W1^T, relu, @W2 -----------
__global__ void __launch_bounds__(32)
k_edge_mlp(const _Float16* __restrict__ h1h, const int* __restrict__ src,
           const int* __restrict__ tgt, const float* __restrict__ ew,
           const _Float16* __restrict__ W1h, const float* __restrict__ w1f,
           const float* __restrict__ b1, const float* __restrict__ w2,
           const float* __restrict__ b2, float* __restrict__ out, int n, int E) {
  long long e2 = (long long)E + 2LL * n;
  int lane = threadIdx.x;
  int c0 = lane & 15, hi = lane >> 4, rowbase = hi * 8;
  long long ntile = (e2 + 15) >> 4;

  v16h bf[4][4];
#pragma unroll
  for (int nt = 0; nt < 4; ++nt)
#pragma unroll
    for (int kc = 0; kc < 4; ++kc)
      bf[nt][kc] = frag_f16(W1h + (size_t)(nt * 16 + c0) * WROW, kc * 32, hi);
  float b1v[4], w1l[4], w2v[4];
#pragma unroll
  for (int nt = 0; nt < 4; ++nt) {
    int col = nt * 16 + c0;
    b1v[nt] = b1[col];
    w1l[nt] = w1f[col * 129 + 128];  // last emb column (edge weight) folded into C-init
    w2v[nt] = w2[col];
  }
  float bias2 = b2[0];

  for (long long tile = blockIdx.x; tile < ntile; tile += gridDim.x) {
    bool full = (tile * 16 + 16 <= e2);          // wave-uniform fast path
    // prefetch next tile's index/weight cachelines (global_prefetch_b8)
    long long tnext = tile + gridDim.x;
    if (tnext * 16 + c0 < E) {
      __builtin_prefetch(&src[tnext * 16 + c0], 0, 0);
      __builtin_prefetch(&tgt[tnext * 16 + c0], 0, 0);
      __builtin_prefetch(&ew[tnext * 16 + c0], 0, 0);
    }

    long long e  = tile * 16 + c0;
    long long ec = e < e2 ? e : e2 - 1;
    int i, jn; long long jj;
    if (ec < E)                     { i = src[ec];           jn = tgt[ec]; jj = jn; }
    else if (ec < (long long)E + n) { i = (int)(ec - E);     jn = i;       jj = (long long)i + n; }
    else                            { i = (int)(ec - E - n); jn = i;       jj = (long long)i + 2LL * n; }

    // ei outputs (ints as f32): branchless on the fast path
    if (full) {
      out[(hi ? e2 : 0) + e] = hi ? (float)jj : (float)i;
    } else if (e < e2) {
      if (hi == 0) out[e] = (float)i;
      else         out[e2 + e] = (float)jj;
    }

    const _Float16* ri = h1h + (size_t)i  * 64;
    const _Float16* rj = h1h + (size_t)jn * 64;
    v16h a0 = frag_f16(ri, 0, hi);
    v16h a1 = frag_f16(ri, 32, hi);
    v16h a2 = frag_f16(rj, 0, hi);
    v16h a3 = frag_f16(rj, 32, hi);

    float wr[8];
    if (tile * 16 + 16 <= E) {                   // uniform: whole tile in real edges
      const float* wp = ew + tile * 16 + rowbase;
      float4 wa = *(const float4*)(wp);
      float4 wb = *(const float4*)(wp + 4);
      wr[0] = wa.x; wr[1] = wa.y; wr[2] = wa.z; wr[3] = wa.w;
      wr[4] = wb.x; wr[5] = wb.y; wr[6] = wb.z; wr[7] = wb.w;
    } else {
#pragma unroll
      for (int r = 0; r < 8; ++r) {
        long long er = tile * 16 + rowbase + r;
        float w;
        if (er >= e2)                   w = 0.0f;
        else if (er < E)                w = ew[er];
        else if (er < (long long)E + n) w = 1.0f;
        else                            w = -1.0f;
        wr[r] = w;
      }
    }

    float partial[8];
#pragma unroll
    for (int r = 0; r < 8; ++r) partial[r] = 0.0f;

#pragma unroll
    for (int nt = 0; nt < 4; ++nt) {
      v8f acc;
#pragma unroll
      for (int r = 0; r < 8; ++r) acc[r] = b1v[nt] + wr[r] * w1l[nt];
      acc = __builtin_amdgcn_wmma_f32_16x16x32_f16(false, a0, false, bf[nt][0], (short)0, acc, false, false);
      acc = __builtin_amdgcn_wmma_f32_16x16x32_f16(false, a1, false, bf[nt][1], (short)0, acc, false, false);
      acc = __builtin_amdgcn_wmma_f32_16x16x32_f16(false, a2, false, bf[nt][2], (short)0, acc, false, false);
      acc = __builtin_amdgcn_wmma_f32_16x16x32_f16(false, a3, false, bf[nt][3], (short)0, acc, false, false);
#pragma unroll
      for (int r = 0; r < 8; ++r) partial[r] += fmaxf(acc[r], 0.0f) * w2v[nt];
    }

    // sum over the 16 columns held by each half-wave (xor<16 stays in the half)
#pragma unroll
    for (int m = 1; m <= 8; m <<= 1)
#pragma unroll
      for (int r = 0; r < 8; ++r) partial[r] += __shfl_xor(partial[r], m, 32);

    if (c0 == 0) {
      float* op = out + 2 * e2 + tile * 16 + rowbase;
      if (full) {
#pragma unroll
        for (int r = 0; r < 8; ++r) op[r] = partial[r] + bias2;
      } else {
#pragma unroll
        for (int r = 0; r < 8; ++r) {
          long long er = tile * 16 + rowbase + r;
          if (er < e2) out[2 * e2 + er] = partial[r] + bias2;
        }
      }
    }
  }
}

extern "C" void kernel_launch(void* const* d_in, const int* in_sizes, int n_in,
                              void* d_out, int out_size, void* d_ws, size_t ws_size,
                              hipStream_t stream) {
  (void)n_in; (void)out_size; (void)ws_size;
  const float* x      = (const float*)d_in[0];
  const int*   eidx   = (const int*)d_in[1];
  const float* ew     = (const float*)d_in[2];
  const float* g0root = (const float*)d_in[3];
  const float* g0rel  = (const float*)d_in[4];
  const float* g0b    = (const float*)d_in[5];
  const float* g1root = (const float*)d_in[6];
  const float* g1rel  = (const float*)d_in[7];
  const float* g1b    = (const float*)d_in[8];
  const float* w1     = (const float*)d_in[9];
  const float* b1     = (const float*)d_in[10];
  const float* w2     = (const float*)d_in[11];
  const float* b2     = (const float*)d_in[12];
  int n = in_sizes[0] / 4;
  int E = in_sizes[2];
  const int* src = eidx;
  const int* tgt = eidx + E;
  float* out = (float*)d_out;
  long long e2 = (long long)E + 2LL * n;

  // workspace carve (256B aligned)
  char* base = (char*)d_ws;
  size_t off = 0;
  auto carve = [&](size_t bytes) -> char* {
    char* p = base + off;
    off += (bytes + 255) & ~(size_t)255;
    return p;
  };
  float*     agg0 = (float*)carve((size_t)n * 4  * sizeof(float));
  float*     agg1 = (float*)carve((size_t)n * 64 * sizeof(float));
  float*     h0   = (float*)carve((size_t)n * 64 * sizeof(float));
  _Float16*  h1h  = (_Float16*)carve((size_t)n * 64 * sizeof(_Float16));
  _Float16*  W1h  = (_Float16*)carve((size_t)64 * WROW * sizeof(_Float16));
  _Float16*  Wg   = (_Float16*)carve((size_t)64 * WROW * sizeof(_Float16));

  {
    long long tot = (long long)n * 64;
    k_init<<<(int)((tot + 255) / 256), 256, 0, stream>>>(agg0, agg1, n, out, e2);
  }
  k_prep<<<(64 * WROW + 255) / 256, 256, 0, stream>>>(w1, g1root, g1rel, W1h, Wg);
  k_gc0_scatter<<<(E + 255) / 256, 256, 0, stream>>>(x, src, tgt, ew, agg0, E);
  k_gc0_node<<<(int)(((long long)n * 64 + 255) / 256), 256, 0, stream>>>(x, agg0, g0root, g0rel, g0b, h0, n);
  k_gc1_scatter<<<(int)(((long long)E * 32 + 255) / 256), 256, 0, stream>>>(h0, src, tgt, ew, agg1, E);
  k_gc1_node<<<2048, 32, 0, stream>>>(h0, agg1, Wg, g1b, h1h, n);
  k_edge_mlp<<<8192, 32, 0, stream>>>(h1h, src, tgt, ew, W1h, w1, b1, w2, b2, out, n, E);
}
